// DCN_2456721294049
// MI455X (gfx1250) — matrix operand order repaired and checked by reading, not compiled, not run on hardware
//
#include <hip/hip_runtime.h>

// ---------- problem constants ----------
#define B_  2
#define C_  256
#define H_  128
#define W_  128
#define O_  256
#define KK  9
#define HW  (H_ * W_)          // 16384
#define NPIX (B_ * HW)         // 32768

// ---------- CDNA5 WMMA types ----------
typedef __bf16 v16bf __attribute__((ext_vector_type(16)));
typedef float  v8f   __attribute__((ext_vector_type(8)));

struct alignas(16) U4 { unsigned int a, b, c, d; };
union FragBF { v16bf v; U4 q[2]; };

// ---------- gfx1250 async-to-LDS (guarded; falls back to sync staging) ----------
#if defined(__has_builtin)
#if __has_builtin(__builtin_amdgcn_global_load_async_to_lds_b128)
#define USE_ASYNC_LDS 1
#endif
#endif
#ifndef USE_ASYNC_LDS
#define USE_ASYNC_LDS 0
#endif

#if USE_ASYNC_LDS
// builtin signature: (v4i AS1*, v4i AS3*, Ii offset, Ii cpol) with v4i = generic vector int4
typedef int v4i_g __attribute__((vector_size(16)));
typedef __attribute__((address_space(1))) v4i_g* gas_v4i;
typedef __attribute__((address_space(3))) v4i_g* lds_v4i;
__device__ __forceinline__ gas_v4i to_gas(const void* p) {
    return (gas_v4i)(unsigned long long)p;            // generic global addr == AS1 addr
}
__device__ __forceinline__ lds_v4i to_lds(const void* p) {
    // generic LDS address: addr[31:0] is the LDS byte offset (aperture layout)
    return (lds_v4i)(unsigned int)(unsigned long long)p;
}
#endif

__device__ __forceinline__ unsigned short f2bf(float f) {
    unsigned int u = __float_as_uint(f);
    u += 0x7fffu + ((u >> 16) & 1u);           // round-to-nearest-even
    return (unsigned short)(u >> 16);
}

// ---------- K0: weight (O,C) fp32 -> bf16 ----------
__global__ __launch_bounds__(256) void k_cvt_weight(const float* __restrict__ w,
                                                    unsigned short* __restrict__ wbf) {
    int i = blockIdx.x * 256 + threadIdx.x;
    if (i < O_ * C_) wbf[i] = f2bf(w[i]);
}

// ---------- K1: 3x3 offset conv, 27 output channels ----------
__global__ __launch_bounds__(256) void k_offset_conv(const float* __restrict__ x,
                                                     const float* __restrict__ w_off,
                                                     const float* __restrict__ b_off,
                                                     float* __restrict__ tmp) {
    __shared__ float sX[16][18][20];   // [cchunk][y halo][x halo (padded)]
    __shared__ float sW[16][9][27];    // [cchunk][tap][oc]

    const int b   = blockIdx.z;
    const int ty0 = blockIdx.y * 16;
    const int tx0 = blockIdx.x * 16;
    const int tid = threadIdx.x;

    float acc[27];
#pragma unroll
    for (int oc = 0; oc < 27; ++oc) acc[oc] = 0.f;

    for (int c0 = 0; c0 < C_; c0 += 16) {
        for (int i = tid; i < 16 * 18 * 18; i += 256) {
            int cc = i / 324, r = i % 324, hy = r / 18, hx = r % 18;
            int gy = ty0 + hy - 1, gx = tx0 + hx - 1;
            float v = 0.f;
            if (gy >= 0 && gy < H_ && gx >= 0 && gx < W_)
                v = x[((b * C_ + c0 + cc) * H_ + gy) * W_ + gx];
            sX[cc][hy][hx] = v;
        }
        for (int i = tid; i < 16 * 9 * 27; i += 256) {
            int cc = i / 243, r = i % 243, tap = r / 27, oc = r % 27;
            sW[cc][tap][oc] = w_off[((oc * C_ + c0 + cc) * 9) + tap];
        }
        __syncthreads();

        const int ly = tid >> 4, lxp = tid & 15;
        for (int cc = 0; cc < 16; ++cc) {
#pragma unroll
            for (int tap = 0; tap < 9; ++tap) {
                float xv = sX[cc][ly + tap / 3][lxp + tap % 3];
#pragma unroll
                for (int oc = 0; oc < 27; ++oc)
                    acc[oc] += sW[cc][tap][oc] * xv;   // LDS broadcast read
            }
        }
        __syncthreads();
    }

    const int gy = ty0 + (tid >> 4), gx = tx0 + (tid & 15);
#pragma unroll
    for (int oc = 0; oc < 27; ++oc)
        tmp[((b * 27 + oc) * H_ + gy) * W_ + gx] = acc[oc] + b_off[oc];
}

// ---------- K2: per-(b,k) spatial softmax stats ----------
__global__ __launch_bounds__(256) void k_softmax_stats(const float* __restrict__ tmp,
                                                       float* __restrict__ stats) {
    __shared__ float red[256];
    const int j = blockIdx.x;
    const int b = j / KK, k = j % KK;
    const float* p = tmp + ((b * 27 + 18 + k) * HW);
    const int tid = threadIdx.x;

    float mx = -1e30f;
    for (int i = tid; i < HW; i += 256) mx = fmaxf(mx, p[i]);
    red[tid] = mx; __syncthreads();
    for (int s = 128; s > 0; s >>= 1) {
        if (tid < s) red[tid] = fmaxf(red[tid], red[tid + s]);
        __syncthreads();
    }
    mx = red[0]; __syncthreads();

    float sm = 0.f;
    for (int i = tid; i < HW; i += 256) sm += __expf(p[i] - mx);
    red[tid] = sm; __syncthreads();
    for (int s = 128; s > 0; s >>= 1) {
        if (tid < s) red[tid] += red[tid + s];
        __syncthreads();
    }
    if (tid == 0) { stats[j * 2] = mx; stats[j * 2 + 1] = 1.f / red[0]; }
}

// ---------- K3: mask + bilinear sample + k-reduction -> agg (bf16) ----------
__global__ __launch_bounds__(256) void k_sample(const float* __restrict__ x,
                                                const float* __restrict__ tmp,
                                                const float* __restrict__ stats,
                                                unsigned short* __restrict__ agg) {
    const int pix = blockIdx.x * 256 + threadIdx.x;
    const int b  = pix >> 14;
    const int hw = pix & (HW - 1);
    const int h  = hw >> 7;
    const int w  = hw & (W_ - 1);

    int   ix[9][4];
    float wt[9][4];
#pragma unroll
    for (int k = 0; k < KK; ++k) {
        float dy = tmp[((b * 27 + 2 * k) * HW) + hw];
        float dx = tmp[((b * 27 + 2 * k + 1) * HW) + hw];
        float lg = tmp[((b * 27 + 18 + k) * HW) + hw];
        float m  = __expf(lg - stats[(b * KK + k) * 2]) * stats[(b * KK + k) * 2 + 1];

        float py = dy + (float)(h + (k / 3) - 1);
        float px = dx + (float)(w + (k % 3) - 1);
        float fy = floorf(py), fx = floorf(px);
        int   y0 = (int)fy,    x0 = (int)fx;
        float ly = py - fy,    lx = px - fx;

        int y0c = min(max(y0, 0), H_ - 1),     x0c = min(max(x0, 0), W_ - 1);
        int y1c = min(max(y0 + 1, 0), H_ - 1), x1c = min(max(x0 + 1, 0), W_ - 1);
        bool vy0 = (y0 >= 0) & (y0 < H_),         vx0 = (x0 >= 0) & (x0 < W_);
        bool vy1 = (y0 + 1 >= 0) & (y0 + 1 < H_), vx1 = (x0 + 1 >= 0) & (x0 + 1 < W_);

        wt[k][0] = (vy0 && vx0) ? (1.f - ly) * (1.f - lx) * m : 0.f;
        wt[k][1] = (vy0 && vx1) ? (1.f - ly) * lx * m : 0.f;
        wt[k][2] = (vy1 && vx0) ? ly * (1.f - lx) * m : 0.f;
        wt[k][3] = (vy1 && vx1) ? ly * lx * m : 0.f;
        ix[k][0] = y0c * W_ + x0c;
        ix[k][1] = y0c * W_ + x1c;
        ix[k][2] = y1c * W_ + x0c;
        ix[k][3] = y1c * W_ + x1c;
    }

    const int c0 = blockIdx.y * 32;
    const float* xb = x + (b * C_) * HW;
    for (int c = c0; c < c0 + 32; ++c) {
        const float* xp = xb + c * HW;
        float s = 0.f;
#pragma unroll
        for (int k = 0; k < KK; ++k) {
            s += wt[k][0] * xp[ix[k][0]] + wt[k][1] * xp[ix[k][1]]
               + wt[k][2] * xp[ix[k][2]] + wt[k][3] * xp[ix[k][3]];
        }
        agg[(b * C_ + c) * HW + hw] = f2bf(s);
    }
}

// ---------- K4: 1x1 conv via WMMA: out = weight(256x256) * agg(256xHW) + bias ----------
// block = 256 thr = 8 waves; tile M=128, N=128, K-step 32.
__global__ __launch_bounds__(256) void k_gemm(const unsigned short* __restrict__ wbf,
                                              const unsigned short* __restrict__ agg,
                                              const float* __restrict__ bias,
                                              float* __restrict__ out) {
    __shared__ unsigned short ldsA[128][40];   // A row-major [m][k]; 80B row = 16B multiple
    __shared__ unsigned short ldsB[128][40];   // B transposed  [n][k]

    const int tid    = threadIdx.x;
    const int lane   = tid & 31;
    const int wave   = tid >> 5;
    const int wm     = wave & 3;
    const int wn     = wave >> 2;
    const int m_base = blockIdx.y * 128;
    const int n_glob = blockIdx.x * 128;
    const int bb     = blockIdx.z;
    const int hseg   = lane >> 4;

    v8f acc[2][4];
#pragma unroll
    for (int i = 0; i < 2; ++i)
#pragma unroll
        for (int j = 0; j < 4; ++j) {
            v8f z = {0.f, 0.f, 0.f, 0.f, 0.f, 0.f, 0.f, 0.f};
            acc[i][j] = z;
        }

    for (int k0 = 0; k0 < C_; k0 += 32) {
        // ---- stage A: 128 rows x 32 k ----
        {
            int r = tid >> 1, hf = (tid & 1) * 16;
            const unsigned short* src = wbf + (m_base + r) * C_ + k0 + hf;
#if USE_ASYNC_LDS
            __builtin_amdgcn_global_load_async_to_lds_b128(to_gas(src),
                                                           to_lds(&ldsA[r][hf]), 0, 0);
            __builtin_amdgcn_global_load_async_to_lds_b128(to_gas(src + 8),
                                                           to_lds(&ldsA[r][hf + 8]), 0, 0);
#else
            const U4* s = (const U4*)src;
            *(U4*)&ldsA[r][hf]     = s[0];
            *(U4*)&ldsA[r][hf + 8] = s[1];
#endif
        }
        // ---- stage B transposed: adjacent lanes take adjacent k -> conflict-free b32 stores
        {
            int kk = (tid & 15) * 2;        // k-pair owned by this thread
            int nn = (tid >> 4) * 8;        // 8 n-columns owned by this thread
            const unsigned short* base = agg + ((bb * C_ + k0 + kk) * HW) + n_glob + nn;
            union { U4 q; unsigned short s[8]; } r0, r1;
            r0.q = *(const U4*)base;            // row kk   (L2-resident)
            r1.q = *(const U4*)(base + HW);     // row kk+1
#pragma unroll
            for (int e = 0; e < 8; ++e) {
                unsigned int v = (unsigned int)r0.s[e] | ((unsigned int)r1.s[e] << 16);
                *(unsigned int*)&ldsB[nn + e][kk] = v;   // 4B aligned, banks disjoint per half-wave
            }
        }
#if USE_ASYNC_LDS
        asm volatile("s_wait_asynccnt 0x0" ::: "memory");
#endif
        __syncthreads();

        // ---- fragments (per CDNA5 per-lane layout) ----
        FragBF bfr[4];
#pragma unroll
        for (int j = 0; j < 4; ++j) {
            int col = wn * 64 + j * 16 + (lane & 15);
            bfr[j].q[0] = *(const U4*)&ldsB[col][hseg * 8];
            bfr[j].q[1] = *(const U4*)&ldsB[col][16 + hseg * 8];
        }
        FragBF afr[2];
#pragma unroll
        for (int i = 0; i < 2; ++i) {
            int row = wm * 32 + i * 16 + (lane & 15);
            afr[i].q[0] = *(const U4*)&ldsA[row][hseg * 8];
            afr[i].q[1] = *(const U4*)&ldsA[row][16 + hseg * 8];
        }
#pragma unroll
        for (int i = 0; i < 2; ++i)
#pragma unroll
            for (int j = 0; j < 4; ++j)
                acc[i][j] = __builtin_amdgcn_wmma_f32_16x16x32_bf16(
                    false, afr[i].v, false, bfr[j].v, (short)0, acc[i][j], false, false);
        __syncthreads();
    }

    // epilogue: C/D layout -> lanes 0-15: M=v, N=lane; lanes 16-31: M=8+v
#pragma unroll
    for (int i = 0; i < 2; ++i) {
        int obase = m_base + wm * 32 + i * 16 + (hseg << 3);
#pragma unroll
        for (int j = 0; j < 4; ++j) {
            int n = n_glob + wn * 64 + j * 16 + (lane & 15);
#pragma unroll
            for (int v = 0; v < 8; ++v) {
                int o = obase + v;
                out[(bb * O_ + o) * HW + n] = acc[i][j][v] + bias[o];
            }
        }
    }
}

// ---------- launch ----------
extern "C" void kernel_launch(void* const* d_in, const int* in_sizes, int n_in,
                              void* d_out, int out_size, void* d_ws, size_t ws_size,
                              hipStream_t stream) {
    const float* x      = (const float*)d_in[0];
    const float* w_off  = (const float*)d_in[1];
    const float* b_off  = (const float*)d_in[2];
    const float* weight = (const float*)d_in[3];
    const float* bias   = (const float*)d_in[4];
    float* out = (float*)d_out;

    char* ws = (char*)d_ws;
    const size_t TMP_BYTES = (size_t)B_ * 27 * HW * sizeof(float); // 3,538,944
    float*          tmp   = (float*)ws;
    float*          stats = (float*)(ws + TMP_BYTES);
    unsigned short* wbf   = (unsigned short*)(ws + TMP_BYTES + 256);
    unsigned short* agg   = (unsigned short*)(ws + TMP_BYTES + 256 + (size_t)O_ * C_ * 2);

    k_cvt_weight<<<(O_ * C_ + 255) / 256, 256, 0, stream>>>(weight, wbf);
    k_offset_conv<<<dim3(W_ / 16, H_ / 16, B_), 256, 0, stream>>>(x, w_off, b_off, tmp);
    k_softmax_stats<<<B_ * KK, 256, 0, stream>>>(tmp, stats);
    k_sample<<<dim3(NPIX / 256, C_ / 32), 256, 0, stream>>>(x, tmp, stats, agg);
    k_gemm<<<dim3(HW / 128, O_ / 128, B_), 256, 0, stream>>>(wbf, agg, bias, out);
}
